// CausalSelfAttention_3186865733989
// MI455X (gfx1250) — matrix operand order repaired and checked by reading, not compiled, not run on hardware
//
#include <hip/hip_runtime.h>
#include <hip/hip_bf16.h>

typedef __attribute__((ext_vector_type(16))) __bf16   v16bf;
typedef __attribute__((ext_vector_type(8)))  float    v8f;
typedef __attribute__((ext_vector_type(4)))  unsigned uv4;
typedef __attribute__((ext_vector_type(8)))  int      iv8;
typedef __attribute__((ext_vector_type(4)))  int      iv4;

#define B_    2
#define T_    4096
#define C_    768
#define H_    12
#define D_    64
#define QKV_N (3 * C_)   // 2304

#if __has_builtin(__builtin_amdgcn_tensor_load_to_lds) && \
    __has_builtin(__builtin_amdgcn_s_wait_tensorcnt)
#define HAS_TDM 1
#else
#define HAS_TDM 0
#endif

// Pack two floats into two bf16 in one dword (RNE via v_cvt)
__device__ __forceinline__ unsigned pack2(float a, float b) {
    union { __bf16 h[2]; unsigned u; } p;
    p.h[0] = (__bf16)a; p.h[1] = (__bf16)b;
    return p.u;
}

// Build a 16-element bf16 fragment from two 16-byte chunks (-> 2x b128 loads)
__device__ __forceinline__ v16bf ld_frag2(const __bf16* p0, const __bf16* p1) {
    union { v16bf v; uint4 u4[2]; } f;
    f.u4[0] = *(const uint4*)p0;
    f.u4[1] = *(const uint4*)p1;
    return f.v;
}

#if HAS_TDM
// TDM 2-D tile load: tile_dim1 rows of tile_dim0 elements (2B each), row
// stride stride0 elements, packed row-major into LDS at lds_off.
// D# fields per CDNA5 ISA ch.8 (group0: count/lds/global/type, group1: dims).
// This toolchain's builtin takes 6 args: (g0, g1, g2, g3, g4ext, cpol).
__device__ __forceinline__ void tdm_load_2d(unsigned lds_off, const void* gaddr,
                                            unsigned dim0, unsigned dim1,
                                            unsigned stride0,
                                            unsigned tile0, unsigned tile1) {
    unsigned long long ga = (unsigned long long)gaddr;
    uv4 g0;
    g0[0] = 1u;                                     // count=1, user mode
    g0[1] = lds_off;                                // lds_addr (bytes)
    g0[2] = (unsigned)(ga & 0xffffffffu);           // global_addr[31:0]
    g0[3] = (unsigned)((ga >> 32) & 0x01ffffffu)    // global_addr[56:32]
            | (2u << 30);                           // type=2 ("image")
    iv8 g1;
    g1[0] = (int)(1u << 16);                        // data_size=1 (2 bytes)
    g1[1] = (int)((dim0 & 0xffffu) << 16);          // tensor_dim0[15:0]
    g1[2] = (int)((dim0 >> 16) | ((dim1 & 0xffffu) << 16));
    g1[3] = (int)((dim1 >> 16) | ((tile0 & 0xffffu) << 16));
    g1[4] = (int)(tile1 & 0xffffu);                 // tile_dim1; tile_dim2=0
    g1[5] = (int)stride0;                           // tensor_dim0_stride lo
    g1[6] = 0;                                      // stride0 hi / stride1 lo
    g1[7] = 0;
    iv4 z4 = {};                                    // 2-D: groups 2/3 unused
    iv8 z8 = {};                                    // extended group unused
    __builtin_amdgcn_tensor_load_to_lds(g0, g1, z4, z4, z8, 0);
}
#endif

// ---------------------------------------------------------------------------
// Kernel 1: QKV projection.  X[B*T, C] fp32 @ W[C, 3C] fp32 -> bf16
//   Q, K laid out [B,H,T,D];  V laid out TRANSPOSED [B,H,D,T].
//   Q pre-scaled by 1/sqrt(D).
// Block: 128 threads (4 waves).  Tile 64(M) x 128(N), K-step 32.
// Each wave owns 64M x 32N: 8 WMMAs per K-step off 12 b128 LDS reads.
// ---------------------------------------------------------------------------
__global__ __launch_bounds__(128)
void qkv_gemm_kernel(const float* __restrict__ X, const float* __restrict__ W,
                     __bf16* __restrict__ Q, __bf16* __restrict__ Kb,
                     __bf16* __restrict__ V)
{
    __shared__ __align__(16) __bf16 As[64 * 32];    // [m][k]
    __shared__ __align__(16) __bf16 Bt[128 * 32];   // [n][k] transposed

    const int tid  = threadIdx.x;
    const int lane = tid & 31;
    const int wv   = tid >> 5;
    const int lh   = lane >> 4;
    const int ln   = lane & 15;

    const int m0 = blockIdx.x * 64;    // row tile in [0, B*T)
    const int n0 = blockIdx.y * 128;   // col tile in [0, 3C)

    v8f acc[2][4] = {};

    for (int k0 = 0; k0 < C_; k0 += 32) {
        // As: 512 float4 units (4 consecutive k, one m) -> packed b64 stores
        #pragma unroll
        for (int i = 0; i < 4; ++i) {
            int u = tid + i * 128;                  // 0..511
            int m = u >> 3, kc = (u & 7) * 4;
            float4 f = *(const float4*)&X[(size_t)(m0 + m) * C_ + k0 + kc];
            uint2 st; st.x = pack2(f.x, f.y); st.y = pack2(f.z, f.w);
            *(uint2*)(As + m * 32 + kc) = st;
        }
        // Bt: 512 units of 2k x 4n over a 32k x 128n tile
        #pragma unroll
        for (int i = 0; i < 4; ++i) {
            int u = tid + i * 128;                  // 0..511
            int kp = u >> 5, n4 = (u & 31) * 4;
            const float* w0 = &W[(size_t)(k0 + 2 * kp) * QKV_N + n0 + n4];
            float4 a = *(const float4*)w0;
            float4 b = *(const float4*)(w0 + QKV_N);
            ((unsigned*)Bt)[((n4 + 0) * 32 + 2 * kp) >> 1] = pack2(a.x, b.x);
            ((unsigned*)Bt)[((n4 + 1) * 32 + 2 * kp) >> 1] = pack2(a.y, b.y);
            ((unsigned*)Bt)[((n4 + 2) * 32 + 2 * kp) >> 1] = pack2(a.z, b.z);
            ((unsigned*)Bt)[((n4 + 3) * 32 + 2 * kp) >> 1] = pack2(a.w, b.w);
        }
        if (k0 + 32 < C_) {                         // prefetch next K-slab
            __builtin_prefetch(&X[(size_t)(m0 + (tid >> 1)) * C_ + k0 + 32], 0, 1);
            __builtin_prefetch(&W[(size_t)(k0 + 32 + (tid >> 2)) * QKV_N + n0], 0, 1);
        }
        __syncthreads();

        v16bf bfr[2];
        #pragma unroll
        for (int ni = 0; ni < 2; ++ni) {
            const __bf16* bp = Bt + (wv * 32 + ni * 16 + ln) * 32 + lh * 8;
            bfr[ni] = ld_frag2(bp, bp + 16);
        }
        #pragma unroll
        for (int mi = 0; mi < 4; ++mi) {
            const __bf16* ap = As + (mi * 16 + ln) * 32;
            v16bf afr = ld_frag2(ap + lh * 8, ap + 16 + lh * 8);
            #pragma unroll
            for (int ni = 0; ni < 2; ++ni)
                acc[ni][mi] = __builtin_amdgcn_wmma_f32_16x16x32_bf16(
                    false, afr, false, bfr[ni], (short)0, acc[ni][mi], false, false);
        }
        __syncthreads();
    }

    // scatter to Q/K (row-major) and V (transposed)
    #pragma unroll
    for (int ni = 0; ni < 2; ++ni) {
        const int c     = n0 + wv * 32 + ni * 16 + ln;  // 0..2303
        const int which = c / C_;                       // 0=q 1=k 2=v
        const int rem   = c - which * C_;
        const int h = rem >> 6, d = rem & 63;
        __bf16* dst = (which == 0) ? Q : (which == 1) ? Kb : V;
        const float scl = (which == 0) ? 0.125f : 1.0f;   // 64^-0.5 into Q
        #pragma unroll
        for (int mi = 0; mi < 4; ++mi) {
            #pragma unroll
            for (int r = 0; r < 8; ++r) {
                int row = m0 + mi * 16 + r + 8 * lh;       // = b*T + t
                int b = row >> 12, t = row & (T_ - 1);
                size_t idx = (which == 2)
                    ? ((size_t)(b * H_ + h) * D_ + d) * T_ + t   // V:[B,H,D,T]
                    : ((size_t)(b * H_ + h) * T_ + t) * D_ + d;  // Q,K
                dst[idx] = (__bf16)(acc[ni][mi][r] * scl);
            }
        }
    }
}

// ---------------------------------------------------------------------------
// Kernel 2: fused causal flash attention.  One block per (b,h, 64-query tile).
// K tile [key][d] and V tile [d][key] staged into LDS by the Tensor Data
// Mover (wave 0 issues 2 descriptors, s_wait_tensorcnt, workgroup barrier).
// All WMMA fragment reads are 2x ds_load_b128.
// ---------------------------------------------------------------------------
__global__ __launch_bounds__(128)
void flash_attn_kernel(const __bf16* __restrict__ Q, const __bf16* __restrict__ Kb,
                       const __bf16* __restrict__ V, __bf16* __restrict__ O)
{
    __shared__ __align__(16) __bf16 Ks[64 * 64];       // [key][d]
    __shared__ __align__(16) __bf16 Vs[64 * 64];       // [d][key]
    __shared__ __align__(16) __bf16 Ps[4][16 * 32];    // per-wave P repack buf

    const int tid  = threadIdx.x;
    const int lane = tid & 31;
    const int wv   = tid >> 5;
    const int lh   = lane >> 4;
    const int ln   = lane & 15;

    const int qblk = blockIdx.x;                       // 0..T/64-1
    const int bh   = blockIdx.y;                       // 0..B*H-1
    const size_t base = (size_t)bh * T_ * D_;

    // Preload Q A-fragments: k(e) = 16*(e>=8) + 8*lh + (e&7)  -> 2x b128 each
    const int qrow_l = qblk * 64 + wv * 16 + ln;
    const __bf16* qp = Q + base + (size_t)qrow_l * D_;
    v16bf qf[2];
    #pragma unroll
    for (int s = 0; s < 2; ++s)
        qf[s] = ld_frag2(qp + s * 32 + lh * 8, qp + s * 32 + 16 + lh * 8);

    v8f o[4] = {};
    float rmax[8], rsum[8];
    #pragma unroll
    for (int r = 0; r < 8; ++r) { rmax[r] = -1e30f; rsum[r] = 0.0f; }

    const int qrow0 = qblk * 64 + wv * 16;             // + r + 8*lh per elem

    for (int jt = 0; jt <= qblk; ++jt) {
        __syncthreads();
#if HAS_TDM
        if (wv == 0) {
            // K tile: 64 rows x 64 d, row stride 64 (flat 8KB)
            tdm_load_2d((unsigned)(size_t)(void*)Ks,
                        Kb + base + (size_t)jt * 64 * D_,
                        64, T_, 64, 64, 64);
            // V tile: 64 d-rows x 64 keys, global row stride T
            tdm_load_2d((unsigned)(size_t)(void*)Vs,
                        V + base + (size_t)jt * 64,
                        T_, D_, T_, 64, 64);
            __builtin_amdgcn_s_wait_tensorcnt(0);
        }
#else
        {
            const uint4* ks = (const uint4*)(Kb + base + (size_t)jt * 64 * D_);
            uint4* kd = (uint4*)Ks;
            #pragma unroll
            for (int i = 0; i < 4; ++i) kd[tid + i * 128] = ks[tid + i * 128];
            #pragma unroll
            for (int i = 0; i < 4; ++i) {
                int idx = tid + i * 128;
                int d = idx >> 3, c8 = (idx & 7) * 8;
                *(uint4*)(Vs + d * 64 + c8) =
                    *(const uint4*)(V + base + (size_t)d * T_ + jt * 64 + c8);
            }
        }
#endif
        __syncthreads();

        #pragma unroll
        for (int half = 0; half < 2; ++half) {         // 32 keys per half
            #pragma unroll
            for (int sub = 0; sub < 2; ++sub) {        // 16-key S sub-tile
                const int kl0 = half * 32 + sub * 16;
                v8f s = {};
                #pragma unroll
                for (int ds = 0; ds < 2; ++ds) {       // contract over D=64
                    const __bf16* kp = Ks + (kl0 + ln) * 64 + ds * 32 + lh * 8;
                    v16bf bfr = ld_frag2(kp, kp + 16);
                    s = __builtin_amdgcn_wmma_f32_16x16x32_bf16(
                        false, qf[ds], false, bfr, (short)0, s, false, false);
                }
                // causal mask + online softmax (row = m, col = key)
                const int kglob = jt * 64 + kl0 + ln;
                float p[8];
                #pragma unroll
                for (int r = 0; r < 8; ++r) {
                    int qg = qrow0 + r + 8 * lh;
                    p[r] = (kglob <= qg) ? s[r] : -1e30f;
                }
                #pragma unroll
                for (int r = 0; r < 8; ++r) {
                    float tm = p[r];                    // row max over 16 keys
                    #pragma unroll
                    for (int m = 1; m < 16; m <<= 1)
                        tm = fmaxf(tm, __shfl_xor(tm, m, 32));
                    float nm    = fmaxf(rmax[r], tm);
                    float alpha = __expf(rmax[r] - nm);
                    float pe    = __expf(p[r] - nm);
                    float ps    = pe;                   // row sum over 16 keys
                    #pragma unroll
                    for (int m = 1; m < 16; m <<= 1)
                        ps += __shfl_xor(ps, m, 32);
                    rsum[r] = rsum[r] * alpha + ps;
                    rmax[r] = nm;
                    #pragma unroll
                    for (int ot = 0; ot < 4; ++ot) o[ot][r] *= alpha;
                    Ps[wv][(r + 8 * lh) * 32 + sub * 16 + ln] = (__bf16)pe;
                }
            }
            // O += P @ V over this half's 32 keys
            const __bf16* pp = Ps[wv] + ln * 32;
            v16bf pf = ld_frag2(pp + lh * 8, pp + 16 + lh * 8);
            #pragma unroll
            for (int ot = 0; ot < 4; ++ot) {           // 4 x 16 cols of D
                const __bf16* vp = Vs + (ot * 16 + ln) * 64 + half * 32 + lh * 8;
                v16bf bfr = ld_frag2(vp, vp + 16);
                o[ot] = __builtin_amdgcn_wmma_f32_16x16x32_bf16(
                    false, pf, false, bfr, (short)0, o[ot], false, false);
            }
        }
    }

    // normalize, store to attn-out [B*T, C] bf16 with col = h*64 + d
    const int b = bh / H_, h = bh - b * H_;
    #pragma unroll
    for (int r = 0; r < 8; ++r) {
        float inv = 1.0f / rsum[r];
        int t = qblk * 64 + wv * 16 + r + 8 * lh;
        size_t row = (size_t)b * T_ + t;
        #pragma unroll
        for (int ot = 0; ot < 4; ++ot)
            O[row * C_ + h * D_ + ot * 16 + ln] = (__bf16)(o[ot][r] * inv);
    }
}

// ---------------------------------------------------------------------------
// Kernel 3: output projection.  A[B*T, C] bf16 @ Wout[C, C] fp32 + bias -> f32
// Tile 64(M) x 128(N), same wave structure as kernel 1.
// ---------------------------------------------------------------------------
__global__ __launch_bounds__(128)
void out_proj_kernel(const __bf16* __restrict__ A, const float* __restrict__ W,
                     const float* __restrict__ bias, float* __restrict__ Y)
{
    __shared__ __align__(16) __bf16 As[64 * 32];    // [m][k]
    __shared__ __align__(16) __bf16 Bt[128 * 32];   // [n][k]

    const int tid  = threadIdx.x;
    const int lane = tid & 31;
    const int wv   = tid >> 5;
    const int lh   = lane >> 4;
    const int ln   = lane & 15;

    const int m0 = blockIdx.x * 64;
    const int n0 = blockIdx.y * 128;

    v8f acc[2][4] = {};

    for (int k0 = 0; k0 < C_; k0 += 32) {
        // As: bf16 source, bulk uint4 copy (256 uint4 units)
        #pragma unroll
        for (int i = 0; i < 2; ++i) {
            int u = tid + i * 128;
            int r = u >> 2, c8 = (u & 3) * 8;
            *(uint4*)(As + r * 32 + c8) =
                *(const uint4*)(A + (size_t)(m0 + r) * C_ + k0 + c8);
        }
        // Bt: transposed fp32 W tile (32k x 128n), packed-pair stores
        #pragma unroll
        for (int i = 0; i < 4; ++i) {
            int u = tid + i * 128;
            int kp = u >> 5, n4 = (u & 31) * 4;
            const float* w0 = &W[(size_t)(k0 + 2 * kp) * C_ + n0 + n4];
            float4 a = *(const float4*)w0;
            float4 b = *(const float4*)(w0 + C_);
            ((unsigned*)Bt)[((n4 + 0) * 32 + 2 * kp) >> 1] = pack2(a.x, b.x);
            ((unsigned*)Bt)[((n4 + 1) * 32 + 2 * kp) >> 1] = pack2(a.y, b.y);
            ((unsigned*)Bt)[((n4 + 2) * 32 + 2 * kp) >> 1] = pack2(a.z, b.z);
            ((unsigned*)Bt)[((n4 + 3) * 32 + 2 * kp) >> 1] = pack2(a.w, b.w);
        }
        if (k0 + 32 < C_) {
            __builtin_prefetch(&A[(size_t)(m0 + (tid >> 1)) * C_ + k0 + 32], 0, 1);
            __builtin_prefetch(&W[(size_t)(k0 + 32 + (tid >> 2)) * C_ + n0], 0, 1);
        }
        __syncthreads();

        v16bf bfr[2];
        #pragma unroll
        for (int ni = 0; ni < 2; ++ni) {
            const __bf16* bp = Bt + (wv * 32 + ni * 16 + ln) * 32 + lh * 8;
            bfr[ni] = ld_frag2(bp, bp + 16);
        }
        #pragma unroll
        for (int mi = 0; mi < 4; ++mi) {
            const __bf16* ap = As + (mi * 16 + ln) * 32;
            v16bf afr = ld_frag2(ap + lh * 8, ap + 16 + lh * 8);
            #pragma unroll
            for (int ni = 0; ni < 2; ++ni)
                acc[ni][mi] = __builtin_amdgcn_wmma_f32_16x16x32_bf16(
                    false, afr, false, bfr[ni], (short)0, acc[ni][mi], false, false);
        }
        __syncthreads();
    }

    #pragma unroll
    for (int ni = 0; ni < 2; ++ni) {
        const int col = n0 + wv * 32 + ni * 16 + ln;
        const float bb = bias[col];
        #pragma unroll
        for (int mi = 0; mi < 4; ++mi) {
            #pragma unroll
            for (int r = 0; r < 8; ++r) {
                int row = m0 + mi * 16 + r + 8 * lh;
                Y[(size_t)row * C_ + col] = acc[ni][mi][r] + bb;
            }
        }
    }
}

// ---------------------------------------------------------------------------
extern "C" void kernel_launch(void* const* d_in, const int* in_sizes, int n_in,
                              void* d_out, int out_size, void* d_ws, size_t ws_size,
                              hipStream_t stream) {
    const float* x     = (const float*)d_in[0];   // [B,T,C]
    const float* w_qkv = (const float*)d_in[1];   // [C,3C]
    const float* w_out = (const float*)d_in[2];   // [C,C]
    const float* b_out = (const float*)d_in[3];   // [C]
    float* out = (float*)d_out;                   // [B,T,C]

    const size_t nqkv = (size_t)B_ * H_ * T_ * D_;  // 6,291,456 elems
    __bf16* Q  = (__bf16*)d_ws;                     // [B,H,T,D]
    __bf16* Kb = Q  + nqkv;                         // [B,H,T,D]
    __bf16* V  = Kb + nqkv;                         // [B,H,D,T]  (transposed)
    __bf16* A2 = V  + nqkv;                         // attn out [B*T, C]
    // total scratch: 4 * 6,291,456 * 2B = 48 MiB

    qkv_gemm_kernel<<<dim3((B_ * T_) / 64, QKV_N / 128), 128, 0, stream>>>(
        x, w_qkv, Q, Kb, V);
    flash_attn_kernel<<<dim3(T_ / 64, B_ * H_), 128, 0, stream>>>(
        Q, Kb, V, A2);
    out_proj_kernel<<<dim3((B_ * T_) / 64, C_ / 128), 128, 0, stream>>>(
        A2, w_out, b_out, out);
}
